// MambaEncoder_23948737643094
// MI455X (gfx1250) — matrix operand order repaired and checked by reading, not compiled, not run on hardware
//
#include <hip/hip_runtime.h>
#include <stdint.h>

// ---------------- problem constants (match reference) ----------------
constexpr int B_ = 4, L_ = 2048, IN_ = 512;
constexpr int D_MODEL = 768, D_INNER = 1536, D_STATE = 64, DT_RANK = 48;
constexpr int D_CONV = 4, DEPTH = 4;
constexpr int PROJ_W = DT_RANK + 2 * D_STATE;   // 176
constexpr int MROWS = B_ * L_;                  // 8192
constexpr float EPS = 1e-6f;

// ---------------- types for WMMA ----------------
typedef __attribute__((ext_vector_type(16))) __bf16 v16bf;
typedef __attribute__((ext_vector_type(8)))  __bf16 v8bf;
typedef __attribute__((ext_vector_type(4)))  __bf16 v4bf;
typedef __attribute__((ext_vector_type(8)))  float  v8f;
typedef __attribute__((ext_vector_type(4)))  float  v4f;

#define DEVFN __device__ __forceinline__

// Native conversions: let clang lower to hardware bf16 cvt ops (RNE).
DEVFN __bf16 f2bf(float f)  { return (__bf16)f; }
DEVFN v4bf   cvt4(v4f f)    { return __builtin_convertvector(f, v4bf); }
DEVFN v8bf   cvt8(v8f f)    { return __builtin_convertvector(f, v8bf); }

DEVFN float sigmoidf_(float x) { return 1.f / (1.f + __expf(-x)); }
DEVFN float siluf_(float x)    { return x * sigmoidf_(x); }
DEVFN float softplusf_(float x){ return (x > 20.f) ? x : log1pf(__expf(x)); }

// ---------------- bf16 WMMA GEMM: C(MxN) = A(MxK) * B(KxN) + epilogue ----------------
// Block: 256 threads = 8 wave32 arranged 4(M) x 2(N); each wave does 2x2 WMMA 16x16
// tiles => block tile 128(M) x 64(N), K-step 32. A/B converted fp32->bf16 into LDS.
// Staging uses a scalar-uniform fast path (unguarded b128 loads / 16B LDS stores);
// the guarded path runs only for the dt_proj K-tail and x_proj's last column block.
enum { EPI_NONE = 0, EPI_EMB = 1, EPI_SOFTPLUS_BIAS = 2, EPI_RESADD = 3 };

template <int EPI>
__global__ __launch_bounds__(256)
void gemm_bf16_wmma(const float* __restrict__ A, int lda,
                    const float* __restrict__ Bw, int ldb,
                    float* __restrict__ C, int ldc,
                    int M, int N, int K,
                    const float* __restrict__ bias,
                    const float* __restrict__ cls_emb,
                    const int*   __restrict__ label,
                    const float* __restrict__ res)
{
  constexpr int LDSA = 40;  // 32 + pad, keeps 16B alignment of fragment reads
  constexpr int LDSB = 40;
  __shared__ __align__(16) __bf16 sA[128 * LDSA];          // [m][k]
  __shared__ __align__(16) __bf16 sB[64 * LDSB];           // [n][k] (transposed)

  const int tid  = threadIdx.x;
  const int lane = tid & 31;
  const int wave = tid >> 5;
  const int wm   = wave & 3;   // 0..3
  const int wn   = wave >> 2;  // 0..1
  const int hi   = lane >> 4;  // half-wave select
  const int l15  = lane & 15;

  const int rowBlk = blockIdx.y * 128;
  const int colBlk = blockIdx.x * 64;
  const bool fullN = (colBlk + 64 <= N);

  v8f acc[2][2];
  const v8f vzero = {0.f, 0.f, 0.f, 0.f, 0.f, 0.f, 0.f, 0.f};
#pragma unroll
  for (int i = 0; i < 2; ++i)
#pragma unroll
    for (int j = 0; j < 2; ++j) acc[i][j] = vzero;

  // staging coordinates (constant across K loop)
  const int ar = tid >> 1;            // A: row 0..127
  const int aks = (tid & 1) * 16;     // A: k-offset 0 or 16
  const int bn = tid & 63;            // B: column within tile
  const int bkc = (tid >> 6) * 8;     // B: k-chunk 0,8,16,24

  const int ktiles = (K + 31) / 32;
  for (int kt = 0; kt < ktiles; ++kt) {
    const int k0 = kt * 32;
    const bool fullK = (k0 + 32 <= K);

    if (fullK) {
      // ---- fast A staging: 4x b128 loads -> packed cvt -> 2x 16B LDS stores ----
      const v4f* ga = (const v4f*)(A + (size_t)(rowBlk + ar) * lda + (k0 + aks));
      const v4f f0 = ga[0], f1 = ga[1], f2 = ga[2], f3 = ga[3];
      const v4bf c0 = cvt4(f0), c1 = cvt4(f1), c2 = cvt4(f2), c3 = cvt4(f3);
      const v8bf v0 = __builtin_shufflevector(c0, c1, 0, 1, 2, 3, 4, 5, 6, 7);
      const v8bf v1 = __builtin_shufflevector(c2, c3, 0, 1, 2, 3, 4, 5, 6, 7);
      __bf16* pa = &sA[ar * LDSA + aks];
      *(v8bf*)pa       = v0;
      *(v8bf*)(pa + 8) = v1;
      if (kt + 1 < ktiles)
        __builtin_prefetch(A + (size_t)(rowBlk + ar) * lda + (k0 + 32 + aks), 0, 1);
    } else {
      // ---- guarded A staging (dt_proj K tail only) ----
      const float* ga = A + (size_t)(rowBlk + ar) * lda + (k0 + aks);
      __bf16* pa = &sA[ar * LDSA + aks];
#pragma unroll
      for (int i = 0; i < 16; ++i) {
        const int kg = k0 + aks + i;
        pa[i] = f2bf((kg < K) ? ga[i] : 0.f);
      }
    }

    if (fullK && fullN) {
      // ---- fast B staging: coalesced column loads, one 16B LDS store ----
      const float* gb = Bw + (size_t)(k0 + bkc) * ldb + (colBlk + bn);
      v8f tmp;
#pragma unroll
      for (int i = 0; i < 8; ++i) tmp[i] = gb[(size_t)i * ldb];
      *(v8bf*)&sB[bn * LDSB + bkc] = cvt8(tmp);
    } else {
      // ---- guarded B staging ----
      const float* gb = Bw + (size_t)(k0 + bkc) * ldb + (colBlk + bn);
      v8f tmp;
#pragma unroll
      for (int i = 0; i < 8; ++i) {
        const int kg = k0 + bkc + i;
        const bool ok = (kg < K) && (colBlk + bn < N);
        tmp[i] = ok ? gb[(size_t)i * ldb] : 0.f;
      }
      *(v8bf*)&sB[bn * LDSB + bkc] = cvt8(tmp);
    }
    __syncthreads();

    // ---- compute: fragments per CDNA5 ISA 7.12.2 bf16 layouts ----
#pragma unroll
    for (int tm = 0; tm < 2; ++tm) {
      // A 16x32: lane m=l15 (both halves), elems 0..7 -> K=hi*8+i,
      //          elems 8..15 -> K=16+hi*8+i
      const int m = wm * 32 + tm * 16 + l15;
      const __bf16* pa = &sA[m * LDSA + hi * 8];
      const v8bf a0 = *(const v8bf*)pa;
      const v8bf a1 = *(const v8bf*)(pa + 16);
      const v16bf afrag = __builtin_shufflevector(
          a0, a1, 0, 1, 2, 3, 4, 5, 6, 7, 8, 9, 10, 11, 12, 13, 14, 15);
#pragma unroll
      for (int tn = 0; tn < 2; ++tn) {
        // B 32x16: lane col n=l15, elems 0..7 -> K = hi*8 + i, elems 8..15 -> K = 16 + hi*8 + i
        const int n = wn * 32 + tn * 16 + l15;
        const __bf16* pb = &sB[n * LDSB + hi * 8];
        const v8bf b0 = *(const v8bf*)pb;
        const v8bf b1 = *(const v8bf*)(pb + 16);
        const v16bf bfrag = __builtin_shufflevector(
            b0, b1, 0, 1, 2, 3, 4, 5, 6, 7, 8, 9, 10, 11, 12, 13, 14, 15);

        acc[tm][tn] = __builtin_amdgcn_wmma_f32_16x16x32_bf16(
            false, afrag, false, bfrag, (short)0, acc[tm][tn], false, false);
      }
    }
    __syncthreads();
  }

  // ---- epilogue + store (C 16x16 f32 layout: VGPR r -> row r + 8*hi, col l15) ----
#pragma unroll
  for (int tm = 0; tm < 2; ++tm) {
#pragma unroll
    for (int tn = 0; tn < 2; ++tn) {
      const int col = colBlk + wn * 32 + tn * 16 + l15;
      if (col < N) {
#pragma unroll
        for (int r = 0; r < 8; ++r) {
          const int row = rowBlk + wm * 32 + tm * 16 + r + hi * 8;
          const size_t off = (size_t)row * ldc + col;
          float v = acc[tm][tn][r];
          if (EPI == EPI_EMB) {
            const int bidx = row / L_;
            v += bias[col] + cls_emb[(size_t)label[bidx] * D_MODEL + col];
          } else if (EPI == EPI_SOFTPLUS_BIAS) {
            v = softplusf_(v + bias[col]);
          } else if (EPI == EPI_RESADD) {
            v += res[off];
          }
          C[off] = v;
        }
      }
    }
  }
}

// ---------------- LayerNorm over D_MODEL, one block per row ----------------
__global__ __launch_bounds__(256)
void layernorm_kernel(const float* __restrict__ x, const float* __restrict__ g,
                      const float* __restrict__ b, float* __restrict__ out)
{
  __shared__ float r1[256], r2[256];
  const int row = blockIdx.x;
  const float* xr = x + (size_t)row * D_MODEL;
  float s = 0.f, ss = 0.f;
  for (int c = threadIdx.x; c < D_MODEL; c += 256) {
    const float v = xr[c]; s += v; ss += v * v;
  }
  r1[threadIdx.x] = s; r2[threadIdx.x] = ss;
  __syncthreads();
  for (int o = 128; o > 0; o >>= 1) {
    if (threadIdx.x < o) { r1[threadIdx.x] += r1[threadIdx.x + o];
                           r2[threadIdx.x] += r2[threadIdx.x + o]; }
    __syncthreads();
  }
  const float mean = r1[0] / D_MODEL;
  const float var  = r2[0] / D_MODEL - mean * mean;
  const float inv  = rsqrtf(var + EPS);
  float* orow = out + (size_t)row * D_MODEL;
  for (int c = threadIdx.x; c < D_MODEL; c += 256)
    orow[c] = (xr[c] - mean) * inv * g[c] + b[c];
}

// ---------------- causal depthwise conv (k=4) + SiLU ----------------
__global__ __launch_bounds__(256)
void conv_silu_kernel(const float* __restrict__ xz, const float* __restrict__ cW,
                      const float* __restrict__ cb, float* __restrict__ xc)
{
  const int idx = blockIdx.x * 256 + threadIdx.x;
  if (idx >= MROWS * D_INNER) return;
  const int d   = idx % D_INNER;
  const int row = idx / D_INNER;      // b*L + l
  const int l   = row % L_;
  float acc = cb[d];
#pragma unroll
  for (int k = 0; k < D_CONV; ++k) {
    const int ls = l + k - (D_CONV - 1);
    if (ls >= 0)
      acc += xz[(size_t)(row + ls - l) * (2 * D_INNER) + d] * cW[d * D_CONV + k];
  }
  xc[idx] = siluf_(acc);
}

// ---------------- sequential selective scan ----------------
// One thread = one (b, d) channel; h[64] state and -exp(A_log) kept in registers.
// Shared B_t / C_t (64+64 floats = 512B) staged per timestep by wave 0 via the
// CDNA5 async-to-LDS path: 32 lanes x global_load_async_to_lds_b128 (16B each),
// completion tracked with ASYNCcnt before the workgroup barrier.
__global__ __launch_bounds__(256)
void scan_kernel(const float* __restrict__ pr, const float* __restrict__ dtv,
                 const float* __restrict__ xc, const float* __restrict__ A_log_l,
                 float* __restrict__ y)
{
  constexpr int CHUNKS = D_INNER / 256;   // 6
  const int b = blockIdx.x / CHUNKS;
  const int d = (blockIdx.x % CHUNKS) * 256 + threadIdx.x;

  float h[D_STATE], a[D_STATE];
#pragma unroll
  for (int n = 0; n < D_STATE; ++n) {
    h[n] = 0.f;
    a[n] = -__expf(A_log_l[(size_t)d * D_STATE + n]);
  }

  __shared__ __align__(16) float sBC[2 * D_STATE];   // [0..63]=B_t, [64..127]=C_t
  const unsigned ldsBase = (unsigned)(size_t)(void*)sBC;

  for (int t = 0; t < L_; ++t) {
    const size_t rb = (size_t)(b * L_ + t);
    const int tid = threadIdx.x;

    if (tid < 32) {
      // lane i copies proj[rb, 48 + 4*i .. 48 + 4*i + 3] -> sBC[4*i ..]
      const float* gp = pr + rb * PROJ_W + DT_RANK + tid * 4;
      const unsigned lds = ldsBase + (unsigned)tid * 16u;
      asm volatile("global_load_async_to_lds_b128 %0, %1, off"
                   :: "v"(lds), "v"((unsigned long long)(size_t)gp)
                   : "memory");
      asm volatile("s_wait_asynccnt 0x0" ::: "memory");
    }
    __syncthreads();

    const float dtt = dtv[rb * D_INNER + d];
    const float xt  = xc[rb * D_INNER + d];
    const float dx  = dtt * xt;
    float acc = 0.f;
#pragma unroll
    for (int n = 0; n < D_STATE; ++n) {
      h[n] = h[n] * __expf(dtt * a[n]) + dx * sBC[n];
      acc += h[n] * sBC[D_STATE + n];
    }
    y[rb * D_INNER + d] = acc;
    __syncthreads();
  }
}

// ---------------- gating: y = (y + xc*D) * silu(z) ----------------
__global__ __launch_bounds__(256)
void gate_kernel(float* __restrict__ y, const float* __restrict__ xc,
                 const float* __restrict__ xz, const float* __restrict__ Dp)
{
  const int idx = blockIdx.x * 256 + threadIdx.x;
  if (idx >= MROWS * D_INNER) return;
  const int d = idx % D_INNER;
  const size_t row = (size_t)(idx / D_INNER);
  const float z = xz[row * (2 * D_INNER) + D_INNER + d];
  y[idx] = (y[idx] + xc[idx] * Dp[d]) * siluf_(z);
}

// ---------------- host side ----------------
extern "C" void kernel_launch(void* const* d_in, const int* in_sizes, int n_in,
                              void* d_out, int out_size, void* d_ws, size_t ws_size,
                              hipStream_t stream)
{
  (void)in_sizes; (void)n_in; (void)out_size; (void)ws_size;
  const float* inputs    = (const float*)d_in[0];
  const int*   label     = (const int*)  d_in[1];
  const float* emb_W     = (const float*)d_in[2];
  const float* emb_b     = (const float*)d_in[3];
  const float* cls_emb   = (const float*)d_in[4];
  const float* norm_g    = (const float*)d_in[5];
  const float* norm_b    = (const float*)d_in[6];
  const float* in_proj_W = (const float*)d_in[7];
  const float* conv_W    = (const float*)d_in[8];
  const float* conv_b    = (const float*)d_in[9];
  const float* x_proj_W  = (const float*)d_in[10];
  const float* dt_proj_W = (const float*)d_in[11];
  const float* dt_proj_b = (const float*)d_in[12];
  const float* A_log     = (const float*)d_in[13];
  const float* Dp        = (const float*)d_in[14];
  const float* out_proj_W= (const float*)d_in[15];

  float* x  = (float*)d_out;            // residual stream lives in d_out
  float* ws = (float*)d_ws;
  const size_t NR = (size_t)MROWS;
  float* xn  = ws;                              // 8192*768
  float* xz  = xn  + NR * D_MODEL;              // 8192*3072
  float* xc  = xz  + NR * (2 * D_INNER);        // 8192*1536
  float* pr  = xc  + NR * D_INNER;              // 8192*176
  float* dtb = pr  + NR * PROJ_W;               // 8192*1536
  float* yb  = dtb + NR * D_INNER;              // 8192*1536

  const dim3 blk(256);
  const int elemTotal = MROWS * D_INNER;
  const dim3 eg((elemTotal + 255) / 256);

  // x = inputs @ emb_W + emb_b + cls_emb[label]
  {
    dim3 g((D_MODEL + 63) / 64, MROWS / 128);
    gemm_bf16_wmma<EPI_EMB><<<g, blk, 0, stream>>>(
        inputs, IN_, emb_W, D_MODEL, x, D_MODEL,
        MROWS, D_MODEL, IN_, emb_b, cls_emb, label, nullptr);
  }

  for (int layer = 0; layer < DEPTH; ++layer) {
    // xn = LayerNorm(x)
    layernorm_kernel<<<MROWS, blk, 0, stream>>>(
        x, norm_g + (size_t)layer * D_MODEL, norm_b + (size_t)layer * D_MODEL, xn);

    // xz = xn @ in_proj_W[layer]   (768 -> 3072)
    {
      dim3 g((2 * D_INNER + 63) / 64, MROWS / 128);
      gemm_bf16_wmma<EPI_NONE><<<g, blk, 0, stream>>>(
          xn, D_MODEL, in_proj_W + (size_t)layer * D_MODEL * 2 * D_INNER, 2 * D_INNER,
          xz, 2 * D_INNER, MROWS, 2 * D_INNER, D_MODEL,
          nullptr, nullptr, nullptr, nullptr);
    }

    // xc = silu(causal_conv(xin) + conv_b)
    conv_silu_kernel<<<eg, blk, 0, stream>>>(
        xz, conv_W + (size_t)layer * D_INNER * D_CONV,
        conv_b + (size_t)layer * D_INNER, xc);

    // pr = xc @ x_proj_W[layer]    (1536 -> 176)
    {
      dim3 g((PROJ_W + 63) / 64, MROWS / 128);
      gemm_bf16_wmma<EPI_NONE><<<g, blk, 0, stream>>>(
          xc, D_INNER, x_proj_W + (size_t)layer * D_INNER * PROJ_W, PROJ_W,
          pr, PROJ_W, MROWS, PROJ_W, D_INNER,
          nullptr, nullptr, nullptr, nullptr);
    }

    // dtb = softplus(pr[:, :48] @ dt_proj_W + dt_proj_b)   (48 -> 1536, K zero-padded)
    {
      dim3 g((D_INNER + 63) / 64, MROWS / 128);
      gemm_bf16_wmma<EPI_SOFTPLUS_BIAS><<<g, blk, 0, stream>>>(
          pr, PROJ_W, dt_proj_W + (size_t)layer * DT_RANK * D_INNER, D_INNER,
          dtb, D_INNER, MROWS, D_INNER, DT_RANK,
          dt_proj_b + (size_t)layer * D_INNER, nullptr, nullptr, nullptr);
    }

    // yb = selective_scan(dtb, pr[B,C], xc, A)
    scan_kernel<<<B_ * (D_INNER / 256), blk, 0, stream>>>(
        pr, dtb, xc, A_log + (size_t)layer * D_INNER * D_STATE, yb);

    // yb = (yb + xc * D) * silu(z)
    gate_kernel<<<eg, blk, 0, stream>>>(yb, xc, xz, Dp + (size_t)layer * D_INNER);

    // x += yb @ out_proj_W[layer]  (1536 -> 768), residual add epilogue
    {
      dim3 g((D_MODEL + 63) / 64, MROWS / 128);
      gemm_bf16_wmma<EPI_RESADD><<<g, blk, 0, stream>>>(
          yb, D_INNER, out_proj_W + (size_t)layer * D_INNER * D_MODEL, D_MODEL,
          x, D_MODEL, MROWS, D_MODEL, D_INNER,
          nullptr, nullptr, nullptr, x);
    }
  }
}
